// AgentAttnModule_4466765988505
// MI455X (gfx1250) — matrix operand order repaired and checked by reading, not compiled, not run on hardware
//
#include <hip/hip_runtime.h>

#define BB 32
#define CC 256
#define HH 56
#define WW 56
#define NN (HH*WW)        // 3136
#define HEADS 8
#define HD 32
#define AG 49
#define SCALE 0.1767766953f   // 32^-0.5

typedef __attribute__((ext_vector_type(16))) _Float16 v16h;
typedef __attribute__((ext_vector_type(8)))  _Float16 v8h;
typedef __attribute__((ext_vector_type(8)))  float    v8f;
typedef __attribute__((ext_vector_type(4)))  int      v4i;

#ifndef __has_builtin
#define __has_builtin(x) 0
#endif
#if __has_builtin(__builtin_amdgcn_global_load_async_to_lds_b128)
#define HAVE_ASYNC_LDS 1
#else
#define HAVE_ASYNC_LDS 0
#endif

typedef __attribute__((address_space(1))) v4i gv4i;   // global int4
typedef __attribute__((address_space(3))) v4i lv4i;   // LDS int4

// 16-byte global -> LDS copy; async (ASYNCcnt) when the toolchain exposes the
// gfx1250 builtin, else a plain vector round-trip through VGPRs.
__device__ __forceinline__ void copy16_g2l(void* lds_dst, const void* gsrc) {
#if HAVE_ASYNC_LDS
  __builtin_amdgcn_global_load_async_to_lds_b128(
      (gv4i*)(uintptr_t)gsrc, (lv4i*)lds_dst, 0, 0);
#else
  *(v8h*)lds_dst = *(const v8h*)gsrc;
#endif
}
__device__ __forceinline__ void copy_wait() {
#if HAVE_ASYNC_LDS
# if __has_builtin(__builtin_amdgcn_s_wait_asynccnt)
  __builtin_amdgcn_s_wait_asynccnt(0);
# else
  asm volatile("s_wait_asynccnt 0" ::: "memory");
# endif
#endif
}

// ---------------------------------------------------------------------------
// WMMA helpers (CDNA5 v_wmma_f32_16x16x32_f16, wave32)
// A operand (16x32 f16), k order chosen here (same as B): lane L holds row (L&15); K = 16*(L>>4)+e
//   => one contiguous 16-half run
// B operand (32x16 f16), k order chosen here (same as A): lane L holds col (L&15); K = 16*(L>>4)+e
//   => one contiguous 16-half run
// C/D (16x16 f32): VGPR r, lane L -> M = r + 8*(L>>4), N = L&15
// ---------------------------------------------------------------------------
__device__ __forceinline__ v16h load_a_frag(const _Float16* row, int lane) {
  return *(const v16h*)(row + (lane >> 4) * 16);
}
__device__ __forceinline__ v8f wmma_f16(v16h a, v16h b, v8f c) {
  return __builtin_amdgcn_wmma_f32_16x16x32_f16(false, a, false, b, (short)0, c,
                                                false, false);
}

// ---------------------------------------------------------------------------
// GEMM: Y[b](256 x 3136) = W(256x256) @ X[b](256x3136), f32 in/out, f16 WMMA.
// block 256 threads (8 waves), tile 128(M) x 64(N), K-steps of 32.
// ---------------------------------------------------------------------------
__global__ __launch_bounds__(256) void gemm_kernel(const float* __restrict__ X,
                                                   const float* __restrict__ Wt,
                                                   float* __restrict__ Y) {
  int b = blockIdx.z;
  int m0 = blockIdx.y * 128;
  int n0 = blockIdx.x * 64;
  int tid = threadIdx.x, lane = tid & 31, wid = tid >> 5;
  int wm = wid & 3, wn = wid >> 2;
  __shared__ __align__(64) _Float16 Ah[128][32];   // [m][k]
  __shared__ __align__(64) _Float16 BhT[64][32];   // [n][k]
  v8f acc00 = {}, acc01 = {}, acc10 = {}, acc11 = {};
  const size_t xbase = (size_t)b * CC * NN;

  for (int kk = 0; kk < 256; kk += 32) {
    {
      // A tile: weights, row-major copy with f32->f16 conversion
      int r = tid >> 1, kh = (tid & 1) * 16;
      const float* ws = Wt + (size_t)(m0 + r) * 256 + kk + kh;
#pragma unroll
      for (int i = 0; i < 16; ++i) Ah[r][kh + i] = (_Float16)ws[i];
      // B tile: transpose-in-registers; lanes sweep the contiguous pixel axis
      // so each of the 8 strided loads is a fully coalesced 128B access, and
      // the LDS write is a single b128 store per thread.
      int nloc = tid & 63, kb = (tid >> 6) * 8;
      const float* xs = X + xbase + (size_t)(kk + kb) * NN + n0 + nloc;
      union { v8h v; _Float16 e[8]; } pk;
#pragma unroll
      for (int i = 0; i < 8; ++i) pk.e[i] = (_Float16)xs[(size_t)i * NN];
      *(v8h*)&BhT[nloc][kb] = pk.v;
      if (kk + 32 < 256)
        __builtin_prefetch(X + xbase + (size_t)(kk + 32 + kb) * NN + n0 + nloc, 0, 1);
    }
    __syncthreads();
    v16h a0 = load_a_frag(&Ah[wm * 32 + (lane & 15)][0], lane);
    v16h a1 = load_a_frag(&Ah[wm * 32 + 16 + (lane & 15)][0], lane);
    v16h b0 = *(const v16h*)&BhT[wn * 32 + (lane & 15)][(lane >> 4) * 16];
    v16h b1 = *(const v16h*)&BhT[wn * 32 + 16 + (lane & 15)][(lane >> 4) * 16];
    acc00 = wmma_f16(a0, b0, acc00);
    acc01 = wmma_f16(a0, b1, acc01);
    acc10 = wmma_f16(a1, b0, acc10);
    acc11 = wmma_f16(a1, b1, acc11);
    __syncthreads();
  }
  float* Yb = Y + (size_t)b * CC * NN;
  int nA = n0 + wn * 32 + (lane & 15);
  int nB = nA + 16;
  int mA = m0 + wm * 32 + 8 * (lane >> 4);
  int mB = mA + 16;
#pragma unroll
  for (int r = 0; r < 8; ++r) {
    Yb[(size_t)(mA + r) * NN + nA] = acc00[r];
    Yb[(size_t)(mA + r) * NN + nB] = acc01[r];
    Yb[(size_t)(mB + r) * NN + nA] = acc10[r];
    Yb[(size_t)(mB + r) * NN + nB] = acc11[r];
  }
}

// ---------------------------------------------------------------------------
// GroupNorm: 32 groups of 8 channels; region per (b,g) is contiguous 8*3136 f32
// ---------------------------------------------------------------------------
__global__ __launch_bounds__(256) void gn_stats_kernel(const float* __restrict__ Y,
                                                       float* __restrict__ stats) {
  int bg = blockIdx.x;                         // b*32+g
  size_t base = (size_t)bg * (8 * NN);
  float s = 0.f, ss = 0.f;
  for (int i = threadIdx.x; i < 8 * NN; i += 256) {
    float v = Y[base + i];
    s += v; ss += v * v;
  }
  __shared__ float rs[256], rss[256];
  rs[threadIdx.x] = s; rss[threadIdx.x] = ss;
  __syncthreads();
  for (int o = 128; o > 0; o >>= 1) {
    if (threadIdx.x < o) { rs[threadIdx.x] += rs[threadIdx.x + o];
                           rss[threadIdx.x] += rss[threadIdx.x + o]; }
    __syncthreads();
  }
  if (threadIdx.x == 0) {
    float m = rs[0] * (1.f / (8 * NN));
    float v = rss[0] * (1.f / (8 * NN)) - m * m;
    stats[bg * 2] = m;
    stats[bg * 2 + 1] = rsqrtf(v + 1e-5f);
  }
}

// apply GN; optionally emit f16 transposed [b][n][c] copy scaled by mul
__global__ __launch_bounds__(256) void gn_apply_kernel(
    const float* __restrict__ Ysrc, float* __restrict__ Ydst,
    const float* __restrict__ gw, const float* __restrict__ gb,
    const float* __restrict__ stats, _Float16* __restrict__ t16, float mul) {
  size_t idx = (size_t)blockIdx.x * 256 + threadIdx.x;   // exactly B*C*N threads
  int n = (int)(idx % NN);
  int c = (int)((idx / NN) % CC);
  int b = (int)(idx / ((size_t)NN * CC));
  int sg = b * 32 + (c >> 3);
  float m = stats[sg * 2], rs = stats[sg * 2 + 1];
  float val = (Ysrc[idx] - m) * rs * gw[c] + gb[c];
  Ydst[idx] = val;
  if (t16) t16[((size_t)b * NN + n) * CC + c] = (_Float16)(val * mul);
}

// ---------------------------------------------------------------------------
// Agent pooling: at = avgpool(q, 7x7); pads agents 49..63 with zeros.
// ---------------------------------------------------------------------------
__global__ __launch_bounds__(256) void pool_kernel(const float* __restrict__ Q,
                                                   _Float16* __restrict__ at_s,
                                                   _Float16* __restrict__ at_u) {
  int idx = blockIdx.x * 256 + threadIdx.x;   // B*8*64*32
  int d = idx & 31, a = (idx >> 5) & 63, h = (idx >> 11) & 7, b = idx >> 14;
  float m = 0.f;
  if (a < AG) {
    int py = a / 7, px = a % 7;
    const float* src = Q + ((size_t)b * CC + h * HD + d) * NN;
    for (int yy = 0; yy < 8; ++yy)
      for (int xx = 0; xx < 8; ++xx)
        m += src[(py * 8 + yy) * WW + px * 8 + xx];
    m *= (1.f / 64.f);
  }
  at_s[idx] = (_Float16)(m * SCALE);
  at_u[idx] = (_Float16)m;
}

// ---------------------------------------------------------------------------
// Positional biases.
// ---------------------------------------------------------------------------
__device__ __forceinline__ float bilin7(const float* m, int y, int x) {
  float sy = (y + 0.5f) * 0.125f - 0.5f;
  float sx = (x + 0.5f) * 0.125f - 0.5f;
  int y0 = (int)floorf(sy); float fy = sy - (float)y0;
  int x0 = (int)floorf(sx); float fx = sx - (float)x0;
  int y0c = min(max(y0, 0), 6), y1c = min(max(y0 + 1, 0), 6);
  int x0c = min(max(x0, 0), 6), x1c = min(max(x0 + 1, 0), 6);
  return (1.f - fy) * ((1.f - fx) * m[y0c * 7 + x0c] + fx * m[y0c * 7 + x1c]) +
         fy * ((1.f - fx) * m[y1c * 7 + x0c] + fx * m[y1c * 7 + x1c]);
}

__global__ __launch_bounds__(256) void bias1_kernel(const float* __restrict__ an,
                                                    const float* __restrict__ ah,
                                                    const float* __restrict__ aw,
                                                    float* __restrict__ pb) {
  int idx = blockIdx.x * 256 + threadIdx.x;
  if (idx >= HEADS * AG * NN) return;
  int n = idx % NN, a = (idx / NN) % AG, h = idx / (NN * AG);
  int y = n / WW, x = n % WW;
  float v = bilin7(an + ((size_t)h * AG + a) * 49, y, x);
  v += ah[((size_t)h * AG + a) * WW + y] + aw[((size_t)h * AG + a) * WW + x];
  pb[idx] = v;
}

__global__ __launch_bounds__(256) void bias2_kernel(const float* __restrict__ na,
                                                    const float* __restrict__ ha,
                                                    const float* __restrict__ wa,
                                                    float* __restrict__ nb2) {
  int idx = blockIdx.x * 256 + threadIdx.x;
  if (idx >= HEADS * NN * 64) return;
  int a = idx & 63, n = (idx >> 6) % NN, h = idx / (NN * 64);
  float v = 0.f;
  if (a < AG) {
    int y = n / WW, x = n % WW;
    v = bilin7(na + ((size_t)h * AG + a) * 49, y, x);
    v += ha[((size_t)h * WW + y) * AG + a] + wa[((size_t)h * WW + x) * AG + a];
  }
  nb2[idx] = v;
}

// ---------------------------------------------------------------------------
// Stage 1: flash attention, agents(49,pad64) attend over N=3136 keys.
// One block per (b,h); online softmax over 49 chunks of 64 columns.
// ---------------------------------------------------------------------------
__global__ __launch_bounds__(256) void stage1_kernel(
    const _Float16* __restrict__ at_s, const _Float16* __restrict__ k16,
    const _Float16* __restrict__ v16, const float* __restrict__ pb,
    _Float16* __restrict__ avT) {
  int bh = blockIdx.x, b = bh >> 3, h = bh & 7;
  int tid = threadIdx.x, lane = tid & 31, wid = tid >> 5;
  int rw = wid & 3, dw = wid >> 2;
  __shared__ __align__(64) _Float16 At[64][32];     // agents [a][d], pre-scaled
  __shared__ __align__(64) _Float16 Kt[64][32];     // keys  [pixel][d]
  __shared__ __align__(64) _Float16 VtT[32][64];    // vals  [d][pixel]
  __shared__ __align__(64) _Float16 Ph[64][64];     // exp(S-m) f16
  __shared__ float S[64][64];
  __shared__ float mrow[64], lrow[64], crow[64];

  {
    int p = tid >> 2, db = (tid & 3) * 8;
    copy16_g2l(&At[p][db],
               at_s + (((size_t)b * HEADS + h) * 64 + p) * 32 + db);
  }
  if (tid < 64) { mrow[tid] = -1e30f; lrow[tid] = 0.f; }
  v8f acc = {};
  const size_t kvbase = (size_t)b * NN * CC + h * HD;
  copy_wait();
  __syncthreads();

  for (int ch = 0; ch < 49; ++ch) {
    int n0 = ch * 64;
    {
      // keys: straight 16B copy (async path)
      int p = tid >> 2, db = (tid & 3) * 8;
      copy16_g2l(&Kt[p][db], k16 + kvbase + (size_t)(n0 + p) * CC + db);
      // values: transpose-in-registers, lanes sweep d (contiguous) so the 8
      // strided loads are coalesced; single b128 LDS store per thread.
      int d = tid & 31, pbase = (tid >> 5) * 8;
      const _Float16* vs = v16 + kvbase + (size_t)(n0 + pbase) * CC + d;
      union { v8h v; _Float16 e[8]; } pk;
#pragma unroll
      for (int i = 0; i < 8; ++i) pk.e[i] = vs[(size_t)i * CC];
      *(v8h*)&VtT[d][pbase] = pk.v;
    }
    copy_wait();
    __syncthreads();
    // S = (at*scale) @ K^T + pb : 16 tiles over 8 waves
#pragma unroll
    for (int tt = 0; tt < 2; ++tt) {
      int t = wid * 2 + tt, ti = t >> 2, tj = t & 3;
      v16h a = load_a_frag(&At[ti * 16 + (lane & 15)][0], lane);
      v16h bb = *(const v16h*)&Kt[tj * 16 + (lane & 15)][(lane >> 4) * 16];
      v8f s = {};
      s = wmma_f16(a, bb, s);
      int col = tj * 16 + (lane & 15);
#pragma unroll
      for (int r = 0; r < 8; ++r) {
        int row = ti * 16 + r + 8 * (lane >> 4);
        float bias = (row < AG) ? pb[((size_t)h * AG + row) * NN + n0 + col] : 0.f;
        S[row][col] = s[r] + bias;
      }
    }
    __syncthreads();
    if (tid < 64) {
      float cm = -1e30f;
#pragma unroll 8
      for (int c = 0; c < 64; ++c) cm = fmaxf(cm, S[tid][c]);
      float mn = fmaxf(mrow[tid], cm);
      float corr = __expf(mrow[tid] - mn);
      float ls = 0.f;
#pragma unroll 8
      for (int c = 0; c < 64; ++c) {
        float p = __expf(S[tid][c] - mn);
        Ph[tid][c] = (_Float16)p;
        ls += p;
      }
      lrow[tid] = lrow[tid] * corr + ls;
      mrow[tid] = mn;
      crow[tid] = corr;
    }
    __syncthreads();
    // rescale accumulator, then acc += P @ V  (K = 64 pixels = 2 wmma k-steps)
#pragma unroll
    for (int r = 0; r < 8; ++r) acc[r] *= crow[rw * 16 + r + 8 * (lane >> 4)];
#pragma unroll
    for (int ks = 0; ks < 2; ++ks) {
      v16h a = load_a_frag(&Ph[rw * 16 + (lane & 15)][ks * 32], lane);
      v16h bb =
          *(const v16h*)&VtT[dw * 16 + (lane & 15)][ks * 32 + (lane >> 4) * 16];
      acc = wmma_f16(a, bb, acc);
    }
    __syncthreads();
  }
  {
    int d = dw * 16 + (lane & 15);
#pragma unroll
    for (int r = 0; r < 8; ++r) {
      int a = rw * 16 + r + 8 * (lane >> 4);
      float v = (a < AG) ? acc[r] / lrow[a] : 0.f;
      avT[(((size_t)b * HEADS + h) * 32 + d) * 64 + a] = (_Float16)v;
    }
  }
}

// ---------------------------------------------------------------------------
// Stage 2: queries attend over 49 agents; xo = P @ agent_v written straight
// into feature-map layout.
// ---------------------------------------------------------------------------
__global__ __launch_bounds__(256) void stage2_kernel(
    const _Float16* __restrict__ q16, const _Float16* __restrict__ at_u,
    const _Float16* __restrict__ avT, const float* __restrict__ nb2,
    float* __restrict__ xo) {
  int blk = blockIdx.x;
  int nt = blk % 49, h = (blk / 49) & 7, b = blk / (49 * HEADS);
  int n0 = nt * 64;
  int tid = threadIdx.x, lane = tid & 31, wid = tid >> 5;
  __shared__ __align__(64) _Float16 Qt[64][32];   // q*scale [pixel][d]
  __shared__ __align__(64) _Float16 Au[64][32];   // agents  [a][d]
  __shared__ __align__(64) _Float16 Av[32][64];   // agent_v [d][a]
  __shared__ __align__(64) _Float16 Ph[64][64];
  __shared__ float S[64][64];
  __shared__ float lrow[64];
  {
    int p = tid >> 2, db = (tid & 3) * 8;
    copy16_g2l(&Qt[p][db],
               q16 + ((size_t)b * NN + n0 + p) * CC + h * HD + db);
    int i = tid * 8;
    size_t hb = (size_t)(b * HEADS + h) * 2048;
    copy16_g2l(&Au[0][0] + i, at_u + hb + i);
    copy16_g2l(&Av[0][0] + i, avT + hb + i);
  }
  copy_wait();
  __syncthreads();
#pragma unroll
  for (int tt = 0; tt < 2; ++tt) {
    int t = wid * 2 + tt, ti = t >> 2, tj = t & 3;
    v16h a = load_a_frag(&Qt[ti * 16 + (lane & 15)][0], lane);
    v16h bb = *(const v16h*)&Au[tj * 16 + (lane & 15)][(lane >> 4) * 16];
    v8f s = {};
    s = wmma_f16(a, bb, s);
    int col = tj * 16 + (lane & 15);
#pragma unroll
    for (int r = 0; r < 8; ++r) {
      int row = ti * 16 + r + 8 * (lane >> 4);
      S[row][col] = (col < AG)
          ? s[r] + nb2[((size_t)h * NN + n0 + row) * 64 + col]
          : -1e30f;
    }
  }
  __syncthreads();
  if (tid < 64) {
    float m = -1e30f;
#pragma unroll 8
    for (int c = 0; c < 64; ++c) m = fmaxf(m, S[tid][c]);
    float ls = 0.f;
#pragma unroll 8
    for (int c = 0; c < 64; ++c) {
      float p = __expf(S[tid][c] - m);
      Ph[tid][c] = (_Float16)p;
      ls += p;
    }
    lrow[tid] = ls;
  }
  __syncthreads();
  int rw = wid & 3, dw = wid >> 2;
  v8f acc = {};
#pragma unroll
  for (int ks = 0; ks < 2; ++ks) {
    v16h a = load_a_frag(&Ph[rw * 16 + (lane & 15)][ks * 32], lane);
    v16h bb = *(const v16h*)&Av[dw * 16 + (lane & 15)][ks * 32 + (lane >> 4) * 16];
    acc = wmma_f16(a, bb, acc);
  }
  int c = h * HD + dw * 16 + (lane & 15);
#pragma unroll
  for (int r = 0; r < 8; ++r) {
    int row = rw * 16 + r + 8 * (lane >> 4);
    xo[((size_t)b * CC + c) * NN + n0 + row] = acc[r] / lrow[row];
  }
}

// ---------------------------------------------------------------------------
// Depthwise 3x3 + bias, accumulated into xo feature map.
// ---------------------------------------------------------------------------
__global__ __launch_bounds__(256) void dwconv_kernel(const float* __restrict__ V,
                                                     const float* __restrict__ wt,
                                                     const float* __restrict__ bias,
                                                     float* __restrict__ XO) {
  size_t idx = (size_t)blockIdx.x * 256 + threadIdx.x;
  int n = (int)(idx % NN);
  int c = (int)((idx / NN) % CC);
  int b = (int)(idx / ((size_t)NN * CC));
  int y = n / WW, x = n % WW;
  const float* w9 = wt + c * 9;
  const float* src = V + ((size_t)b * CC + c) * NN;
  float s = bias[c];
#pragma unroll
  for (int dy = -1; dy <= 1; ++dy) {
    int yy = y + dy;
    if ((unsigned)yy >= HH) continue;
#pragma unroll
    for (int dx = -1; dx <= 1; ++dx) {
      int xx = x + dx;
      if ((unsigned)xx >= WW) continue;
      s += src[yy * WW + xx] * w9[(dy + 1) * 3 + dx + 1];
    }
  }
  XO[idx] += s;
}

// ---------------------------------------------------------------------------
// Host launcher
// ---------------------------------------------------------------------------
extern "C" void kernel_launch(void* const* d_in, const int* in_sizes, int n_in,
                              void* d_out, int out_size, void* d_ws, size_t ws_size,
                              hipStream_t stream) {
  (void)in_sizes; (void)n_in; (void)out_size; (void)ws_size;
  const float* x    = (const float*)d_in[0];
  const float* q_w  = (const float*)d_in[1];
  const float* q_gw = (const float*)d_in[2];
  const float* q_gb = (const float*)d_in[3];
  const float* k_w  = (const float*)d_in[4];
  const float* k_gw = (const float*)d_in[5];
  const float* k_gb = (const float*)d_in[6];
  const float* v_w  = (const float*)d_in[7];
  const float* v_gw = (const float*)d_in[8];
  const float* v_gb = (const float*)d_in[9];
  const float* p_w  = (const float*)d_in[10];
  const float* p_gw = (const float*)d_in[11];
  const float* p_gb = (const float*)d_in[12];
  const float* dwc_w = (const float*)d_in[13];
  const float* dwc_b = (const float*)d_in[14];
  const float* an_b = (const float*)d_in[15];
  const float* na_b = (const float*)d_in[16];
  const float* ah_b = (const float*)d_in[17];
  const float* aw_b = (const float*)d_in[18];
  const float* ha_b = (const float*)d_in[19];
  const float* wa_b = (const float*)d_in[20];

  const size_t fmapF = (size_t)BB * CC * NN;          // 25,690,112 floats
  char* w = (char*)d_ws;
  float* S0  = (float*)w; w += fmapF * 4;             // gemm scratch (q,k,p)
  float* Vf  = (float*)w; w += fmapF * 4;             // normalized v (f32)
  float* XO  = (float*)w; w += fmapF * 4;             // attn output feature map
  _Float16* q16 = (_Float16*)w; w += fmapF * 2;       // [b][n][c], *scale
  _Float16* k16 = (_Float16*)w; w += fmapF * 2;       // [b][n][c]
  _Float16* v16 = (_Float16*)w; w += fmapF * 2;       // [b][n][c]
  float* pb  = (float*)w; w += (size_t)HEADS * AG * NN * 4;
  float* nb2 = (float*)w; w += (size_t)HEADS * NN * 64 * 4;
  _Float16* at_s = (_Float16*)w; w += (size_t)BB * HEADS * 64 * 32 * 2;
  _Float16* at_u = (_Float16*)w; w += (size_t)BB * HEADS * 64 * 32 * 2;
  _Float16* avT  = (_Float16*)w; w += (size_t)BB * HEADS * 32 * 64 * 2;
  float* stats = (float*)w; w += (size_t)BB * 32 * 2 * 4;

  dim3 gGemm(49, 2, BB);
  const int elemBlocks = (int)(fmapF / 256);          // 100352

  // positional biases (independent)
  bias1_kernel<<<(HEADS * AG * NN + 255) / 256, 256, 0, stream>>>(an_b, ah_b, aw_b, pb);
  bias2_kernel<<<(HEADS * NN * 64 + 255) / 256, 256, 0, stream>>>(na_b, ha_b, wa_b, nb2);

  // q = GN(x @ q_w); f16 copy carries the HD^-1/2 scale for stage-2
  gemm_kernel<<<gGemm, 256, 0, stream>>>(x, q_w, S0);
  gn_stats_kernel<<<BB * 32, 256, 0, stream>>>(S0, stats);
  gn_apply_kernel<<<elemBlocks, 256, 0, stream>>>(S0, S0, q_gw, q_gb, stats, q16, SCALE);
  pool_kernel<<<(BB * HEADS * 64 * 32) / 256, 256, 0, stream>>>(S0, at_s, at_u);

  // k (reuse S0)
  gemm_kernel<<<gGemm, 256, 0, stream>>>(x, k_w, S0);
  gn_stats_kernel<<<BB * 32, 256, 0, stream>>>(S0, stats);
  gn_apply_kernel<<<elemBlocks, 256, 0, stream>>>(S0, S0, k_gw, k_gb, stats, k16, 1.f);

  // v (kept in Vf for the depthwise residual)
  gemm_kernel<<<gGemm, 256, 0, stream>>>(x, v_w, Vf);
  gn_stats_kernel<<<BB * 32, 256, 0, stream>>>(Vf, stats);
  gn_apply_kernel<<<elemBlocks, 256, 0, stream>>>(Vf, Vf, v_gw, v_gb, stats, v16, 1.f);

  // two-stage agent attention
  stage1_kernel<<<BB * HEADS, 256, 0, stream>>>(at_s, k16, v16, pb, avT);
  stage2_kernel<<<BB * HEADS * 49, 256, 0, stream>>>(q16, at_u, avT, nb2, XO);
  dwconv_kernel<<<elemBlocks, 256, 0, stream>>>(Vf, dwc_w, dwc_b, XO);

  // final projection + GN into d_out
  gemm_kernel<<<gGemm, 256, 0, stream>>>(XO, p_w, S0);
  gn_stats_kernel<<<BB * 32, 256, 0, stream>>>(S0, stats);
  gn_apply_kernel<<<elemBlocks, 256, 0, stream>>>(S0, (float*)d_out, p_gw, p_gb,
                                                  stats, nullptr, 1.f);
}
